// DoubleStreamBlock_73005854098041
// MI455X (gfx1250) — matrix operand order repaired and checked by reading, not compiled, not run on hardware
//
#include <hip/hip_runtime.h>

// ---------------------------------------------------------------------------
// CDNA5 (gfx1250) double-stream block. wave32, WMMA bf16 16x16x32, f32 accum.
// ---------------------------------------------------------------------------

typedef __attribute__((ext_vector_type(16))) __bf16 v16bf;
typedef __attribute__((ext_vector_type(8)))  float  v8f;
typedef int v4i_vs __attribute__((vector_size(16)));   // matches builtin proto

union BF16x16 { v16bf v; uint4 u[2]; };
union Pack4   { __bf16 h[4]; uint2 u; };

#define HIDN    3072
#define NHEAD   24
#define HDIM    128
#define MLPD    12288
#define L_IMG   1024
#define L_TXT   512
#define L_TOT   1536
#define QKVW    9216   // 3*HIDN

// async global->LDS (ASYNCcnt path); guarded so either toolchain compiles
#if defined(__HIP_DEVICE_COMPILE__) && defined(__has_builtin)
# if __has_builtin(__builtin_amdgcn_global_load_async_to_lds_b128)
#  define USE_ASYNC_LDS 1
# endif
#endif
#ifndef USE_ASYNC_LDS
# define USE_ASYNC_LDS 0
#endif

#if USE_ASYNC_LDS
static __device__ __forceinline__ void async_copy_b128(const void* g, void* l) {
    __builtin_amdgcn_global_load_async_to_lds_b128(
        (__attribute__((address_space(1))) v4i_vs*)g,
        (__attribute__((address_space(3))) v4i_vs*)l, 0, 0);
}
static __device__ __forceinline__ void wait_async() {
# if __has_builtin(__builtin_amdgcn_s_wait_asynccnt)
    __builtin_amdgcn_s_wait_asynccnt(0);
# else
    asm volatile("s_wait_asynccnt 0" ::: "memory");
# endif
}
#endif

static __device__ __forceinline__ v8f wmma_bf16(v16bf a, v16bf b, v8f c) {
    return __builtin_amdgcn_wmma_f32_16x16x32_bf16(false, a, false, b, (short)0, c,
                                                   false, false);
}

static __device__ __forceinline__ float gelu_tanh(float x) {
    float x3 = x * x * x;
    return 0.5f * x * (1.0f + tanhf(0.7978845608028654f * (x + 0.044715f * x3)));
}

// ---------------------------------------------------------------------------
// 1) mod vectors: out[stream][6*HIDN] = silu(vec) @ mod_w + mod_b
// ---------------------------------------------------------------------------
__global__ __launch_bounds__(256) void k_mod(const float* __restrict__ vec,
                                             const float* __restrict__ wimg,
                                             const float* __restrict__ bimg,
                                             const float* __restrict__ wtxt,
                                             const float* __restrict__ btxt,
                                             float* __restrict__ mod) {
    __shared__ float sv[HIDN];
    int t = threadIdx.x;
    for (int i = t; i < HIDN; i += 256) {
        float x = vec[i];
        sv[i] = x / (1.0f + __expf(-x));
    }
    __syncthreads();
    int g      = blockIdx.x * 256 + t;        // 0 .. 2*18432-1
    int stream = g / (6 * HIDN);
    int col    = g % (6 * HIDN);
    const float* W = stream ? wtxt : wimg;
    const float* B = stream ? btxt : bimg;
    float acc = 0.0f;
    for (int k = 0; k < HIDN; k += 4) {
        acc += sv[k + 0] * W[(size_t)(k + 0) * (6 * HIDN) + col];
        acc += sv[k + 1] * W[(size_t)(k + 1) * (6 * HIDN) + col];
        acc += sv[k + 2] * W[(size_t)(k + 2) * (6 * HIDN) + col];
        acc += sv[k + 3] * W[(size_t)(k + 3) * (6 * HIDN) + col];
    }
    mod[(size_t)stream * (6 * HIDN) + col] = acc + B[col];
}

// ---------------------------------------------------------------------------
// 2) layernorm + modulation, f32 -> bf16:  out = (1+scale)*LN(x) + shift
// ---------------------------------------------------------------------------
__global__ __launch_bounds__(256) void k_lnmod(const float* __restrict__ x,
                                               const float* __restrict__ shift,
                                               const float* __restrict__ scale,
                                               __bf16* __restrict__ out) {
    int row = blockIdx.x;
    int t   = threadIdx.x;
    const float* xr = x + (size_t)row * HIDN;
    float s = 0.0f, s2 = 0.0f;
    for (int i = t; i < HIDN; i += 256) {
        float v = xr[i];
        s += v; s2 += v * v;
    }
    #pragma unroll
    for (int m = 16; m > 0; m >>= 1) {
        s  += __shfl_xor(s,  m, 32);
        s2 += __shfl_xor(s2, m, 32);
    }
    __shared__ float rs[8], rs2[8];
    if ((t & 31) == 0) { rs[t >> 5] = s; rs2[t >> 5] = s2; }
    __syncthreads();
    float tot = 0.0f, tot2 = 0.0f;
    #pragma unroll
    for (int i = 0; i < 8; ++i) { tot += rs[i]; tot2 += rs2[i]; }
    float mean = tot * (1.0f / HIDN);
    float var  = tot2 * (1.0f / HIDN) - mean * mean;
    float rr   = rsqrtf(var + 1e-6f);
    __bf16* orow = out + (size_t)row * HIDN;
    for (int i = t; i < HIDN; i += 256) {
        float v = (xr[i] - mean) * rr;
        orow[i] = (__bf16)((1.0f + scale[i]) * v + shift[i]);
    }
}

// ---------------------------------------------------------------------------
// 3) WMMA GEMM: C[M,N] = A_bf16[M,K] @ W_f32->bf16[K,N] (+epilogue)
//    block tile 64x128, 8 waves (2x4), wave tile 32x32, K-step 32,
//    double-buffered LDS, one barrier per K-step, software-pipelined staging.
//    EPI: 0 = f32 out, 1 = gelu->bf16, 2 = resid + gate*(c+bias) -> f32
// ---------------------------------------------------------------------------
template <int EPI>
__global__ __launch_bounds__(256) void k_gemm(const __bf16* __restrict__ A, int lda,
                                              const float* __restrict__ W,
                                              const float* __restrict__ bias,
                                              int M, int N, int K,
                                              float* __restrict__ outf,
                                              __bf16* __restrict__ outb,
                                              const float* __restrict__ gate,
                                              const float* __restrict__ resid) {
    __shared__ __align__(16) __bf16 As[2][64 * 48];
    __shared__ __align__(16) __bf16 Ws[2][128 * 48];

    int t    = threadIdx.x;
    int bx   = blockIdx.x, by = blockIdx.y;
    int lane = t & 31, wid = t >> 5;
    int wm   = wid >> 2, wn = wid & 3;      // 2 x 4 waves
    int tm0  = wm * 32, tn0 = wn * 32;
    int half = lane >> 4, l16 = lane & 15;

    v8f acc[2][2];
    #pragma unroll
    for (int i = 0; i < 2; ++i)
        #pragma unroll
        for (int j = 0; j < 2; ++j)
            acc[i][j] = (v8f){0, 0, 0, 0, 0, 0, 0, 0};

    // A tile 64x32 bf16: one uint4 per thread
    int arow = t >> 2, ac0 = (t & 3) * 8;
    // W tile 32x128 f32: 4k x 4n fp32 sub-block per thread
    int kq = (t >> 5) * 4, nq = (t & 31) * 4;

    const __bf16* Ag = A + (size_t)(by * 64 + arow) * lda + ac0;
    const float*  Wg = W + (size_t)kq * N + (size_t)bx * 128 + nq;

    float4 wr0, wr1, wr2, wr3;
#if !USE_ASYNC_LDS
    uint4 areg;
#endif
    // prologue: stage tile 0
#if USE_ASYNC_LDS
    async_copy_b128(Ag, &As[0][arow * 48 + ac0]);
#else
    areg = *(const uint4*)Ag;
#endif
    wr0 = *(const float4*)(Wg);
    wr1 = *(const float4*)(Wg + (size_t)N);
    wr2 = *(const float4*)(Wg + 2 * (size_t)N);
    wr3 = *(const float4*)(Wg + 3 * (size_t)N);

    const int niter = K >> 5;
    for (int i = 0; i < niter; ++i) {
        int cur = i & 1;
        // W regs -> LDS transposed (Ws[n][k]): 4x ds_store_b64, packed bf16 quads
        float wk[4][4];
        *(float4*)&wk[0][0] = wr0;
        *(float4*)&wk[1][0] = wr1;
        *(float4*)&wk[2][0] = wr2;
        *(float4*)&wk[3][0] = wr3;
        #pragma unroll
        for (int c = 0; c < 4; ++c) {
            Pack4 p;
            p.h[0] = (__bf16)wk[0][c];
            p.h[1] = (__bf16)wk[1][c];
            p.h[2] = (__bf16)wk[2][c];
            p.h[3] = (__bf16)wk[3][c];
            *(uint2*)&Ws[cur][(nq + c) * 48 + kq] = p.u;
        }
#if USE_ASYNC_LDS
        wait_async();                       // As[cur] resident
#else
        *(uint4*)&As[cur][arow * 48 + ac0] = areg;
#endif
        __syncthreads();

        // stage next tile while computing this one
        if (i + 1 < niter) {
            Ag += 32;
            Wg += (size_t)32 * N;
#if USE_ASYNC_LDS
            async_copy_b128(Ag, &As[1 - cur][arow * 48 + ac0]);
#else
            areg = *(const uint4*)Ag;
#endif
            __builtin_prefetch(Wg + 32 * (size_t)N, 0, 1);
            wr0 = *(const float4*)(Wg);
            wr1 = *(const float4*)(Wg + (size_t)N);
            wr2 = *(const float4*)(Wg + 2 * (size_t)N);
            wr3 = *(const float4*)(Wg + 3 * (size_t)N);
        }

        BF16x16 af[2], bf[2];
        #pragma unroll
        for (int mi = 0; mi < 2; ++mi) {
            int mr = tm0 + mi * 16 + l16;
            af[mi].u[0] = *(const uint4*)&As[cur][mr * 48 + half * 8];
            af[mi].u[1] = *(const uint4*)&As[cur][mr * 48 + 16 + half * 8];
        }
        #pragma unroll
        for (int ni = 0; ni < 2; ++ni) {
            int nc = tn0 + ni * 16 + l16;
            bf[ni].u[0] = *(const uint4*)&Ws[cur][nc * 48 + half * 8];
            bf[ni].u[1] = *(const uint4*)&Ws[cur][nc * 48 + 16 + half * 8];
        }
        #pragma unroll
        for (int mi = 0; mi < 2; ++mi)
            #pragma unroll
            for (int ni = 0; ni < 2; ++ni)
                acc[mi][ni] = wmma_bf16(af[mi].v, bf[ni].v, acc[mi][ni]);
    }

    // epilogue (C layout: vgpr r -> row r + 8*half, col = l16)
    #pragma unroll
    for (int mi = 0; mi < 2; ++mi) {
        #pragma unroll
        for (int ni = 0; ni < 2; ++ni) {
            int colbase = bx * 128 + tn0 + ni * 16 + l16;
            float bv = bias[colbase];
            #pragma unroll
            for (int r = 0; r < 8; ++r) {
                int row = by * 64 + tm0 + mi * 16 + r + half * 8;
                float c = acc[mi][ni][r] + bv;
                if constexpr (EPI == 0) {
                    outf[(size_t)row * N + colbase] = c;
                } else if constexpr (EPI == 1) {
                    outb[(size_t)row * N + colbase] = (__bf16)gelu_tanh(c);
                } else {
                    outf[(size_t)row * N + colbase] =
                        resid[(size_t)row * N + colbase] + gate[colbase] * c;
                }
            }
        }
    }
}

// ---------------------------------------------------------------------------
// 4) qkv post: RMS-norm q/k, RoPE, emit Q/K bf16 [h][s][d] and V^T [h][d][s]
// ---------------------------------------------------------------------------
__global__ __launch_bounds__(64) void k_qkv_post(const float* __restrict__ qkv_img,
                                                 const float* __restrict__ qkv_txt,
                                                 const float* __restrict__ qn_img,
                                                 const float* __restrict__ kn_img,
                                                 const float* __restrict__ qn_txt,
                                                 const float* __restrict__ kn_txt,
                                                 const float* __restrict__ pe,
                                                 __bf16* __restrict__ Q,
                                                 __bf16* __restrict__ Kk,
                                                 __bf16* __restrict__ Vt) {
    int h = blockIdx.x;
    int s = blockIdx.y;          // token in concat order: txt[0..511], img[512..]
    int t = threadIdx.x;         // 0..63 -> pair (2t, 2t+1)
    bool istxt = (s < L_TXT);
    const float* row = istxt ? (qkv_txt + (size_t)s * QKVW)
                             : (qkv_img + (size_t)(s - L_TXT) * QKVW);
    const float* qn = istxt ? qn_txt : qn_img;
    const float* kn = istxt ? kn_txt : kn_img;
    int d0 = 2 * t;
    float q0 = row[(0 * NHEAD + h) * HDIM + d0];
    float q1 = row[(0 * NHEAD + h) * HDIM + d0 + 1];
    float k0 = row[(1 * NHEAD + h) * HDIM + d0];
    float k1 = row[(1 * NHEAD + h) * HDIM + d0 + 1];
    float v0 = row[(2 * NHEAD + h) * HDIM + d0];
    float v1 = row[(2 * NHEAD + h) * HDIM + d0 + 1];

    float qs = q0 * q0 + q1 * q1;
    float ks = k0 * k0 + k1 * k1;
    #pragma unroll
    for (int m = 16; m > 0; m >>= 1) {
        qs += __shfl_xor(qs, m, 32);
        ks += __shfl_xor(ks, m, 32);
    }
    __shared__ float sq[2], sk[2];
    if ((t & 31) == 0) { sq[t >> 5] = qs; sk[t >> 5] = ks; }
    __syncthreads();
    float qr = rsqrtf((sq[0] + sq[1]) * (1.0f / HDIM) + 1e-6f);
    float kr = rsqrtf((sk[0] + sk[1]) * (1.0f / HDIM) + 1e-6f);
    q0 = q0 * qr * qn[d0]; q1 = q1 * qr * qn[d0 + 1];
    k0 = k0 * kr * kn[d0]; k1 = k1 * kr * kn[d0 + 1];

    const float* p = pe + ((size_t)s * 64 + t) * 4;   // {cos, -sin, sin, cos}
    float qo0 = p[0] * q0 + p[1] * q1;
    float qo1 = p[2] * q0 + p[3] * q1;
    float ko0 = p[0] * k0 + p[1] * k1;
    float ko1 = p[2] * k0 + p[3] * k1;

    size_t base = ((size_t)h * L_TOT + s) * HDIM + d0;
    Q[base] = (__bf16)qo0;  Q[base + 1] = (__bf16)qo1;
    Kk[base] = (__bf16)ko0; Kk[base + 1] = (__bf16)ko1;
    Vt[((size_t)h * HDIM + d0) * L_TOT + s]     = (__bf16)v0;
    Vt[((size_t)h * HDIM + d0 + 1) * L_TOT + s] = (__bf16)v1;
}

// ---------------------------------------------------------------------------
// 5) flash attention: 1 wave per (32 q-rows, head); K/V fragments shared by
//    both q-subtiles (halves L2 traffic, 32 WMMA per 32-key tile).
// ---------------------------------------------------------------------------
__global__ __launch_bounds__(32) void k_attn(const __bf16* __restrict__ Q,
                                             const __bf16* __restrict__ Kk,
                                             const __bf16* __restrict__ Vt,
                                             __bf16* __restrict__ AO) {
    const float SCALE = 0.08838834764831843f;   // 1/sqrt(128)
    int qt = blockIdx.x, h = blockIdx.y;        // 32 q-rows starting qt*32
    int lane = threadIdx.x;
    int half = lane >> 4, l16 = lane & 15;

    __shared__ __align__(16) __bf16 Pl[32 * 48];

    const __bf16* Qh = Q  + ((size_t)h * L_TOT + qt * 32) * HDIM;
    const __bf16* Kh = Kk + (size_t)h * L_TOT * HDIM;
    const __bf16* Vh = Vt + (size_t)h * HDIM * L_TOT;

    BF16x16 qf[2][4];                    // two q-subtiles, d = 0,32,64,96
    #pragma unroll
    for (int qs = 0; qs < 2; ++qs)
        #pragma unroll
        for (int d = 0; d < 4; ++d) {
            const __bf16* qp = Qh + (size_t)(qs * 16 + l16) * HDIM + d * 32;
            qf[qs][d].u[0] = *(const uint4*)(qp + half * 8);
            qf[qs][d].u[1] = *(const uint4*)(qp + 16 + half * 8);
        }

    v8f o[2][8];
    float mr[2][8], lr[2][8];
    #pragma unroll
    for (int qs = 0; qs < 2; ++qs)
        #pragma unroll
        for (int i = 0; i < 8; ++i) {
            o[qs][i] = (v8f){0, 0, 0, 0, 0, 0, 0, 0};
            mr[qs][i] = -3.0e38f;
            lr[qs][i] = 0.0f;
        }

    for (int kv0 = 0; kv0 < L_TOT; kv0 += 32) {
        v8f s[2][2];
        #pragma unroll
        for (int qs = 0; qs < 2; ++qs) {
            s[qs][0] = (v8f){0, 0, 0, 0, 0, 0, 0, 0};
            s[qs][1] = (v8f){0, 0, 0, 0, 0, 0, 0, 0};
        }
        #pragma unroll
        for (int d = 0; d < 4; ++d) {      // K frags loaded once, used 2x
            BF16x16 b0, b1;
            const __bf16* kp0 = Kh + (size_t)(kv0 + l16) * HDIM + d * 32 + half * 8;
            const __bf16* kp1 = Kh + (size_t)(kv0 + 16 + l16) * HDIM + d * 32 + half * 8;
            b0.u[0] = *(const uint4*)kp0; b0.u[1] = *(const uint4*)(kp0 + 16);
            b1.u[0] = *(const uint4*)kp1; b1.u[1] = *(const uint4*)(kp1 + 16);
            #pragma unroll
            for (int qs = 0; qs < 2; ++qs) {
                s[qs][0] = wmma_bf16(qf[qs][d].v, b0.v, s[qs][0]);
                s[qs][1] = wmma_bf16(qf[qs][d].v, b1.v, s[qs][1]);
            }
        }
        // online softmax per q-subtile (row r lives in this 16-lane half)
        #pragma unroll
        for (int qs = 0; qs < 2; ++qs) {
            #pragma unroll
            for (int r = 0; r < 8; ++r) {
                float a = s[qs][0][r] * SCALE;
                float b = s[qs][1][r] * SCALE;
                float mx = fmaxf(a, b);
                mx = fmaxf(mx, __shfl_xor(mx, 1, 32));
                mx = fmaxf(mx, __shfl_xor(mx, 2, 32));
                mx = fmaxf(mx, __shfl_xor(mx, 4, 32));
                mx = fmaxf(mx, __shfl_xor(mx, 8, 32));
                float mnew  = fmaxf(mr[qs][r], mx);
                float alpha = __expf(mr[qs][r] - mnew);
                float p0 = __expf(a - mnew);
                float p1 = __expf(b - mnew);
                float sum = p0 + p1;
                sum += __shfl_xor(sum, 1, 32);
                sum += __shfl_xor(sum, 2, 32);
                sum += __shfl_xor(sum, 4, 32);
                sum += __shfl_xor(sum, 8, 32);
                lr[qs][r] = lr[qs][r] * alpha + sum;
                mr[qs][r] = mnew;
                Pl[(qs * 16 + r + half * 8) * 48 + l16]      = (__bf16)p0;
                Pl[(qs * 16 + r + half * 8) * 48 + 16 + l16] = (__bf16)p1;
                #pragma unroll
                for (int dt = 0; dt < 8; ++dt) o[qs][dt][r] *= alpha;
            }
        }
        // P: C-layout -> A-layout via LDS (single wave; compiler inserts waits)
        BF16x16 pf[2];
        #pragma unroll
        for (int qs = 0; qs < 2; ++qs) {
            pf[qs].u[0] = *(const uint4*)&Pl[(qs * 16 + l16) * 48 + half * 8];
            pf[qs].u[1] = *(const uint4*)&Pl[(qs * 16 + l16) * 48 + 16 + half * 8];
        }
        #pragma unroll
        for (int dt = 0; dt < 8; ++dt) {   // V frags loaded once, used 2x
            BF16x16 vf;
            const __bf16* vp = Vh + (size_t)(dt * 16 + l16) * L_TOT + kv0 + half * 8;
            vf.u[0] = *(const uint4*)vp;
            vf.u[1] = *(const uint4*)(vp + 16);
            o[0][dt] = wmma_bf16(pf[0].v, vf.v, o[0][dt]);
            o[1][dt] = wmma_bf16(pf[1].v, vf.v, o[1][dt]);
        }
    }

    #pragma unroll
    for (int qs = 0; qs < 2; ++qs)
        #pragma unroll
        for (int r = 0; r < 8; ++r) {
            float inv = 1.0f / lr[qs][r];
            int srow  = qt * 32 + qs * 16 + r + half * 8;
            #pragma unroll
            for (int dt = 0; dt < 8; ++dt)
                AO[(size_t)srow * HIDN + h * HDIM + dt * 16 + l16] =
                    (__bf16)(o[qs][dt][r] * inv);
        }
}

// ---------------------------------------------------------------------------
// host launcher
// ---------------------------------------------------------------------------
extern "C" void kernel_launch(void* const* d_in, const int* in_sizes, int n_in,
                              void* d_out, int out_size, void* d_ws, size_t ws_size,
                              hipStream_t stream) {
    const float* img      = (const float*)d_in[0];
    const float* txt      = (const float*)d_in[1];
    const float* vec      = (const float*)d_in[2];
    const float* pe       = (const float*)d_in[3];
    const float* img_mod_w = (const float*)d_in[4];
    const float* img_mod_b = (const float*)d_in[5];
    const float* txt_mod_w = (const float*)d_in[6];
    const float* txt_mod_b = (const float*)d_in[7];
    const float* img_qkv_w = (const float*)d_in[8];
    const float* img_qkv_b = (const float*)d_in[9];
    const float* txt_qkv_w = (const float*)d_in[10];
    const float* txt_qkv_b = (const float*)d_in[11];
    const float* img_qnorm = (const float*)d_in[12];
    const float* img_knorm = (const float*)d_in[13];
    const float* txt_qnorm = (const float*)d_in[14];
    const float* txt_knorm = (const float*)d_in[15];
    const float* img_proj_w = (const float*)d_in[16];
    const float* img_proj_b = (const float*)d_in[17];
    const float* txt_proj_w = (const float*)d_in[18];
    const float* txt_proj_b = (const float*)d_in[19];
    const float* img_mlp1_w = (const float*)d_in[20];
    const float* img_mlp1_b = (const float*)d_in[21];
    const float* img_mlp2_w = (const float*)d_in[22];
    const float* img_mlp2_b = (const float*)d_in[23];
    const float* txt_mlp1_w = (const float*)d_in[24];
    const float* txt_mlp1_b = (const float*)d_in[25];
    const float* txt_mlp2_w = (const float*)d_in[26];
    const float* txt_mlp2_b = (const float*)d_in[27];

    // workspace layout (bytes)
    const size_t OFF_XM   = 256 * 1024;                                  // bf16 1536x3072
    const size_t OFF_QKVF = OFF_XM   + (size_t)L_TOT * HIDN * 2;         // f32 1536x9216
    const size_t OFF_QB   = OFF_QKVF + (size_t)L_TOT * QKVW * 4;
    const size_t OFF_KB   = OFF_QB   + (size_t)L_TOT * HIDN * 2;
    const size_t OFF_VT   = OFF_KB   + (size_t)L_TOT * HIDN * 2;
    const size_t OFF_AO   = OFF_VT   + (size_t)L_TOT * HIDN * 2;
    const size_t OFF_X2   = OFF_AO   + (size_t)L_TOT * HIDN * 2;         // f32
    const size_t OFF_HB   = OFF_X2   + (size_t)L_TOT * HIDN * 4;         // bf16 1536x12288

    char* wsb = (char*)d_ws;
    float*  modbuf  = (float*)(wsb);                 // [2][6*3072]
    __bf16* XM      = (__bf16*)(wsb + OFF_XM);       // img rows 0..1023, txt 1024..
    float*  QKVF    = (float*)(wsb + OFF_QKVF);      // img 1024x9216, txt 512x9216
    __bf16* QB      = (__bf16*)(wsb + OFF_QB);
    __bf16* KB      = (__bf16*)(wsb + OFF_KB);
    __bf16* VT      = (__bf16*)(wsb + OFF_VT);
    __bf16* AO      = (__bf16*)(wsb + OFF_AO);       // txt rows 0..511, img 512..
    float*  X2      = (float*)(wsb + OFF_X2);        // img 0..1023, txt 1024..
    __bf16* HB      = (__bf16*)(wsb + OFF_HB);

    float* mod_img = modbuf;
    float* mod_txt = modbuf + 6 * HIDN;
    float* out     = (float*)d_out;                  // img then txt

    // 1) modulation vectors
    k_mod<<<(2 * 6 * HIDN) / 256, 256, 0, stream>>>(vec, img_mod_w, img_mod_b,
                                                    txt_mod_w, txt_mod_b, modbuf);
    // 2) LN+mod -> bf16 activations
    k_lnmod<<<L_IMG, 256, 0, stream>>>(img, mod_img + 0 * HIDN, mod_img + 1 * HIDN, XM);
    k_lnmod<<<L_TXT, 256, 0, stream>>>(txt, mod_txt + 0 * HIDN, mod_txt + 1 * HIDN,
                                       XM + (size_t)L_IMG * HIDN);
    // 3) qkv GEMMs (f32 out)
    k_gemm<0><<<dim3(QKVW / 128, L_IMG / 64), 256, 0, stream>>>(
        XM, HIDN, img_qkv_w, img_qkv_b, L_IMG, QKVW, HIDN, QKVF, nullptr, nullptr, nullptr);
    k_gemm<0><<<dim3(QKVW / 128, L_TXT / 64), 256, 0, stream>>>(
        XM + (size_t)L_IMG * HIDN, HIDN, txt_qkv_w, txt_qkv_b, L_TXT, QKVW, HIDN,
        QKVF + (size_t)L_IMG * QKVW, nullptr, nullptr, nullptr);
    // 4) rms + rope + layout
    k_qkv_post<<<dim3(NHEAD, L_TOT), 64, 0, stream>>>(
        QKVF, QKVF + (size_t)L_IMG * QKVW, img_qnorm, img_knorm, txt_qnorm, txt_knorm,
        pe, QB, KB, VT);
    // 5) attention (32 q-rows per block)
    k_attn<<<dim3(L_TOT / 32, NHEAD), 32, 0, stream>>>(QB, KB, VT, AO);
    // 6) proj + gated residual -> X2
    k_gemm<2><<<dim3(HIDN / 128, L_IMG / 64), 256, 0, stream>>>(
        AO + (size_t)L_TXT * HIDN, HIDN, img_proj_w, img_proj_b, L_IMG, HIDN, HIDN,
        X2, nullptr, mod_img + 2 * HIDN, img);
    k_gemm<2><<<dim3(HIDN / 128, L_TXT / 64), 256, 0, stream>>>(
        AO, HIDN, txt_proj_w, txt_proj_b, L_TXT, HIDN, HIDN,
        X2 + (size_t)L_IMG * HIDN, nullptr, mod_txt + 2 * HIDN, txt);
    // 7) LN+mod (2nd) -> bf16 (reuse XM)
    k_lnmod<<<L_IMG, 256, 0, stream>>>(X2, mod_img + 3 * HIDN, mod_img + 4 * HIDN, XM);
    k_lnmod<<<L_TXT, 256, 0, stream>>>(X2 + (size_t)L_IMG * HIDN,
                                       mod_txt + 3 * HIDN, mod_txt + 4 * HIDN,
                                       XM + (size_t)L_IMG * HIDN);
    // 8) mlp1 + gelu -> bf16
    k_gemm<1><<<dim3(MLPD / 128, L_IMG / 64), 256, 0, stream>>>(
        XM, HIDN, img_mlp1_w, img_mlp1_b, L_IMG, MLPD, HIDN,
        nullptr, HB, nullptr, nullptr);
    k_gemm<1><<<dim3(MLPD / 128, L_TXT / 64), 256, 0, stream>>>(
        XM + (size_t)L_IMG * HIDN, HIDN, txt_mlp1_w, txt_mlp1_b, L_TXT, MLPD, HIDN,
        nullptr, HB + (size_t)L_IMG * MLPD, nullptr, nullptr);
    // 9) mlp2 + gated residual -> d_out (img first, then txt)
    k_gemm<2><<<dim3(HIDN / 128, L_IMG / 64), 256, 0, stream>>>(
        HB, MLPD, img_mlp2_w, img_mlp2_b, L_IMG, HIDN, MLPD,
        out, nullptr, mod_img + 5 * HIDN, X2);
    k_gemm<2><<<dim3(HIDN / 128, L_TXT / 64), 256, 0, stream>>>(
        HB + (size_t)L_IMG * MLPD, MLPD, txt_mlp2_w, txt_mlp2_b, L_TXT, HIDN, MLPD,
        out + (size_t)L_IMG * HIDN, nullptr, mod_txt + 5 * HIDN,
        X2 + (size_t)L_IMG * HIDN);
    (void)in_sizes; (void)n_in; (void)out_size; (void)ws_size;
}
